// BertModelWrapper_zeroout_44418551775568
// MI455X (gfx1250) — matrix operand order repaired and checked by reading, not compiled, not run on hardware
//
#include <hip/hip_runtime.h>

// ---------------------------------------------------------------------------
// BERT-base forward for MI455X (gfx1250, wave32, WMMA).
// Dense GEMMs: LDS-staged 128x128 block tiles (8 waves, 32x64 per wave,
// 8 x v_wmma_f32_16x16x32_bf16 per k-step per wave), double-buffered K=32
// slices, bank-conflict-free padded LDS. Activations + weights staged in bf16
// (weights ~170MB -> L2-resident). fp32 accumulation everywhere.
// ---------------------------------------------------------------------------

constexpr int Lc  = 12;
constexpr int Hc  = 768;
constexpr int NHc = 12;
constexpr int DHc = 64;
constexpr int FFc = 3072;
constexpr int Bc  = 4;
constexpr int Sc  = 512;
constexpr int Mtok = Bc * Sc;  // 2048 tokens

typedef __attribute__((ext_vector_type(16))) __bf16 v16bf;
typedef __attribute__((ext_vector_type(8)))  __bf16 v8bf;
typedef __attribute__((ext_vector_type(8)))  float  v8f;

__device__ __forceinline__ __bf16 f2bf(float f) {
  unsigned u = __builtin_bit_cast(unsigned, f);
  unsigned r = (u + 0x7FFFu + ((u >> 16) & 1u)) >> 16;  // round-to-nearest-even
  unsigned short us = (unsigned short)r;
  return __builtin_bit_cast(__bf16, us);
}

__device__ __forceinline__ v8f wmma_bf16(v16bf a, v16bf b, v8f c) {
  return __builtin_amdgcn_wmma_f32_16x16x32_bf16(
      false, a, false, b, (short)0, c, false, false);
}

__device__ __forceinline__ v16bf pack16(v8bf lo, v8bf hi) {
  v16bf v;
#pragma unroll
  for (int i = 0; i < 8; ++i) { v[i] = lo[i]; v[8 + i] = hi[i]; }
  return v;
}

// A-operand tile 16x32 from bf16 row-major [M][K]. Lane L: row = L&15;
// elems i = K k0+(L>>4)*16+i (contiguous 32B run).
__device__ __forceinline__ v16bf load_a16x32_bf(const __bf16* __restrict__ A, int lda,
                                                int row, int k0, int khalf) {
  const __bf16* ap = A + (size_t)row * lda + k0 + khalf * 16;
  return pack16(*(const v8bf*)(ap), *(const v8bf*)(ap + 8));
}

// B-operand tile 32x16 from bf16 transposed [N][K]. Lane L: col = L&15;
// elems i = K k0+(L>>4)*16+i (contiguous 32B run).
__device__ __forceinline__ v16bf load_b32x16_bf(const __bf16* __restrict__ Bt, int ldb,
                                                int col, int k0, int khalf) {
  const __bf16* bp = Bt + (size_t)col * ldb + k0 + khalf * 16;
  return pack16(*(const v8bf*)(bp), *(const v8bf*)(bp + 8));
}

// ---------------------------------------------------------------------------
// LDS-staged WMMA GEMM. Block: 256 thr = 8 waves -> 128x128 tile.
// Wave (wy 0..3, wx 0..1): 32 rows x 64 cols = 2x4 accumulators.
//   out = act(alpha * Abf[M,K] @ Bt^T + bias + residual)
// C fp32 / Cbf bf16 each nullable. act: 0 none, 1 exact GELU, 2 tanh.
// ---------------------------------------------------------------------------
constexpr int LDSP = 40;  // padded row stride (bf16) -> conflict-free b128 reads

__global__ __launch_bounds__(256) void gemm_wmma_kernel(
    const __bf16* __restrict__ A, int lda,
    const __bf16* __restrict__ Bt, int ldb,
    float* __restrict__ C, __bf16* __restrict__ Cbf, int ldc,
    const float* __restrict__ bias,
    const float* __restrict__ residual,
    int M, int N, int K, float alpha, int act)
{
  __shared__ __bf16 As[2][128][LDSP];
  __shared__ __bf16 Bs[2][128][LDSP];

  int gx = (N + 127) >> 7;
  int bx = blockIdx.x % gx, by = blockIdx.x / gx;
  int blockM = by << 7, blockN = bx << 7;

  int tid  = threadIdx.x;
  int wave = tid >> 5, lane = tid & 31;
  int wy = wave & 3, wx = wave >> 2;        // 4 (M) x 2 (N) wave grid
  int mrow = lane & 15, khalf = lane >> 4;

  // cooperative copy mapping: thread -> (row, 32B half) of the 128x32 slice
  int crow  = tid >> 1, chalf = tid & 1;
  int garow = blockM + crow; if (garow >= M) garow = M - 1;   // clamped reads
  int gbrow = blockN + crow; if (gbrow >= N) gbrow = N - 1;
  const __bf16* gA = A  + (size_t)garow * lda + chalf * 16;
  const __bf16* gB = Bt + (size_t)gbrow * ldb + chalf * 16;

  // prologue: stage K-slice 0 into buffer 0
  {
    v8bf a0 = *(const v8bf*)(gA);
    v8bf a1 = *(const v8bf*)(gA + 8);
    v8bf b0 = *(const v8bf*)(gB);
    v8bf b1 = *(const v8bf*)(gB + 8);
    *(v8bf*)&As[0][crow][chalf * 16]     = a0;
    *(v8bf*)&As[0][crow][chalf * 16 + 8] = a1;
    *(v8bf*)&Bs[0][crow][chalf * 16]     = b0;
    *(v8bf*)&Bs[0][crow][chalf * 16 + 8] = b1;
  }
  __syncthreads();

  v8f acc[2][4] = {};
  int buf = 0;
  for (int k0 = 0; k0 < K; k0 += 32) {
    bool has_next = (k0 + 32) < K;
    v8bf nA0 = {}, nA1 = {}, nB0 = {}, nB1 = {};
    if (has_next) {
      const __bf16* pa = gA + k0 + 32;
      const __bf16* pb = gB + k0 + 32;
      nA0 = *(const v8bf*)(pa); nA1 = *(const v8bf*)(pa + 8);
      nB0 = *(const v8bf*)(pb); nB1 = *(const v8bf*)(pb + 8);
      if (k0 + 64 < K) {
        __builtin_prefetch(gA + k0 + 64, 0, 1);
        __builtin_prefetch(gB + k0 + 64, 0, 1);
      }
    }

    // operands from LDS[buf] (ds_load_b128, padded rows -> conflict-free)
    v16bf at[2], bt[4];
#pragma unroll
    for (int mi = 0; mi < 2; ++mi) {
      const __bf16* p = &As[buf][wy * 32 + mi * 16 + mrow][khalf * 16];
      at[mi] = pack16(*(const v8bf*)(p), *(const v8bf*)(p + 8));
    }
#pragma unroll
    for (int ni = 0; ni < 4; ++ni) {
      const __bf16* p = &Bs[buf][wx * 64 + ni * 16 + mrow][khalf * 16];
      bt[ni] = pack16(*(const v8bf*)(p), *(const v8bf*)(p + 8));
    }
#pragma unroll
    for (int mi = 0; mi < 2; ++mi)
#pragma unroll
      for (int ni = 0; ni < 4; ++ni)
        acc[mi][ni] = wmma_bf16(at[mi], bt[ni], acc[mi][ni]);

    if (has_next) {
      int nb = buf ^ 1;
      *(v8bf*)&As[nb][crow][chalf * 16]     = nA0;
      *(v8bf*)&As[nb][crow][chalf * 16 + 8] = nA1;
      *(v8bf*)&Bs[nb][crow][chalf * 16]     = nB0;
      *(v8bf*)&Bs[nb][crow][chalf * 16 + 8] = nB1;
    }
    __syncthreads();
    buf ^= 1;
  }

  // epilogue
#pragma unroll
  for (int mi = 0; mi < 2; ++mi) {
#pragma unroll
    for (int ni = 0; ni < 4; ++ni) {
      int col = blockN + wx * 64 + ni * 16 + mrow;
      if (col >= N) continue;
      float bb = bias ? bias[col] : 0.f;
      v8f a = acc[mi][ni];
#pragma unroll
      for (int r = 0; r < 8; ++r) {
        int row = blockM + wy * 32 + mi * 16 + r + khalf * 8;
        if (row >= M) continue;
        float v = alpha * a[r] + bb;
        if (residual) v += residual[(size_t)row * ldc + col];
        if (act == 1)      v = 0.5f * v * (1.f + erff(v * 0.70710678118654752f));
        else if (act == 2) v = tanhf(v);
        if (C)   C[(size_t)row * ldc + col]   = v;
        if (Cbf) Cbf[(size_t)row * ldc + col] = f2bf(v);
      }
    }
  }
}

// ---------------------------------------------------------------------------
// Attention scores: P = (q.k^T)/8 + ext, 32x32 wave tile, K=DH=64.
// grid: x = (16*16)/8 = 32, z = B*NH = 48.
// ---------------------------------------------------------------------------
__global__ __launch_bounds__(256) void attn_scores_kernel(
    const __bf16* __restrict__ qbf, const __bf16* __restrict__ kbf,
    const float* __restrict__ ext, float* __restrict__ attn)
{
  int z = blockIdx.z;
  int b = z / NHc, h = z % NHc;
  const __bf16* Aq = qbf + (size_t)b * Sc * Hc + h * DHc;
  const __bf16* Bk = kbf + (size_t)z * Sc * DHc;
  float*        P  = attn + (size_t)z * Sc * Sc;
  const float*  eb = ext + b * Sc;

  int tilesX = Sc / 32;  // 16
  int wid = blockIdx.x * 8 + (threadIdx.x >> 5);
  int ty = wid / tilesX, tx = wid % tilesX;
  int lane = threadIdx.x & 31;
  int mrow = lane & 15, khalf = lane >> 4;

  v8f acc00 = {}, acc01 = {}, acc10 = {}, acc11 = {};
#pragma unroll
  for (int k0 = 0; k0 < DHc; k0 += 32) {
    v16bf a0 = load_a16x32_bf(Aq, Hc, ty * 32 + mrow,      k0, khalf);
    v16bf a1 = load_a16x32_bf(Aq, Hc, ty * 32 + 16 + mrow, k0, khalf);
    v16bf b0 = load_b32x16_bf(Bk, DHc, tx * 32 + mrow,      k0, khalf);
    v16bf b1 = load_b32x16_bf(Bk, DHc, tx * 32 + 16 + mrow, k0, khalf);
    acc00 = wmma_bf16(a0, b0, acc00);
    acc01 = wmma_bf16(a0, b1, acc01);
    acc10 = wmma_bf16(a1, b0, acc10);
    acc11 = wmma_bf16(a1, b1, acc11);
  }
  v8f* accs[4] = { &acc00, &acc01, &acc10, &acc11 };
#pragma unroll
  for (int mi = 0; mi < 2; ++mi) {
#pragma unroll
    for (int ni = 0; ni < 2; ++ni) {
      int col = tx * 32 + ni * 16 + mrow;
      float e = eb[col];
      v8f a = *accs[mi * 2 + ni];
#pragma unroll
      for (int r = 0; r < 8; ++r) {
        int row = ty * 32 + mi * 16 + r + khalf * 8;
        P[(size_t)row * Sc + col] = a[r] * 0.125f + e;
      }
    }
  }
}

// Softmax over rows of 512 (in place) + bf16 copy for the ctx GEMM.
__global__ __launch_bounds__(256) void softmax_kernel(float* __restrict__ attn,
                                                      __bf16* __restrict__ pbf)
{
  __shared__ float red[256];
  float*  rowp = attn + (size_t)blockIdx.x * Sc;
  __bf16* rowb = pbf  + (size_t)blockIdx.x * Sc;
  int tid = threadIdx.x;
  float x0 = rowp[tid], x1 = rowp[tid + 256];
  red[tid] = fmaxf(x0, x1);
  __syncthreads();
  for (int w = 128; w > 0; w >>= 1) { if (tid < w) red[tid] = fmaxf(red[tid], red[tid + w]); __syncthreads(); }
  float m = red[0];
  __syncthreads();
  float e0 = __expf(x0 - m), e1 = __expf(x1 - m);
  red[tid] = e0 + e1;
  __syncthreads();
  for (int w = 128; w > 0; w >>= 1) { if (tid < w) red[tid] += red[tid + w]; __syncthreads(); }
  float inv = 1.f / red[0];
  float p0 = e0 * inv, p1 = e1 * inv;
  rowp[tid] = p0;        rowp[tid + 256] = p1;
  rowb[tid] = f2bf(p0);  rowb[tid + 256] = f2bf(p1);
}

// ctx = probs @ v : 32x32 wave tiles, K=S=512. grid: x=(16*2)/8=4, z=48.
__global__ __launch_bounds__(256) void attn_ctx_kernel(
    const __bf16* __restrict__ pbf, const __bf16* __restrict__ vT,
    float* __restrict__ ctx, __bf16* __restrict__ ctxbf)
{
  int z = blockIdx.z;
  int b = z / NHc, h = z % NHc;
  const __bf16* Ap = pbf + (size_t)z * Sc * Sc;
  const __bf16* Bv = vT  + (size_t)z * DHc * Sc;
  size_t cbase = (size_t)b * Sc * Hc + h * DHc;

  int tilesX = DHc / 32;  // 2
  int wid = blockIdx.x * 8 + (threadIdx.x >> 5);
  int ty = wid / tilesX, tx = wid % tilesX;
  int lane = threadIdx.x & 31;
  int mrow = lane & 15, khalf = lane >> 4;

  v8f acc00 = {}, acc01 = {}, acc10 = {}, acc11 = {};
  for (int k0 = 0; k0 < Sc; k0 += 32) {
    if (k0 + 32 < Sc) {
      __builtin_prefetch(Ap + (size_t)(ty * 32 + mrow) * Sc + k0 + 32, 0, 1);
      __builtin_prefetch(Bv + (size_t)(tx * 32 + mrow) * Sc + k0 + 32, 0, 1);
    }
    v16bf a0 = load_a16x32_bf(Ap, Sc, ty * 32 + mrow,      k0, khalf);
    v16bf a1 = load_a16x32_bf(Ap, Sc, ty * 32 + 16 + mrow, k0, khalf);
    v16bf b0 = load_b32x16_bf(Bv, Sc, tx * 32 + mrow,      k0, khalf);
    v16bf b1 = load_b32x16_bf(Bv, Sc, tx * 32 + 16 + mrow, k0, khalf);
    acc00 = wmma_bf16(a0, b0, acc00);
    acc01 = wmma_bf16(a0, b1, acc01);
    acc10 = wmma_bf16(a1, b0, acc10);
    acc11 = wmma_bf16(a1, b1, acc11);
  }
  v8f* accs[4] = { &acc00, &acc01, &acc10, &acc11 };
#pragma unroll
  for (int mi = 0; mi < 2; ++mi) {
#pragma unroll
    for (int ni = 0; ni < 2; ++ni) {
      int col = tx * 32 + ni * 16 + mrow;
      v8f a = *accs[mi * 2 + ni];
#pragma unroll
      for (int r = 0; r < 8; ++r) {
        int row = ty * 32 + mi * 16 + r + khalf * 8;
        size_t off = cbase + (size_t)row * Hc + col;
        ctx[off]   = a[r];
        ctxbf[off] = f2bf(a[r]);
      }
    }
  }
}

// Weight transpose + fp32->bf16: Wt[n*K + k] = bf16(W[k*N + n]).
__global__ void wt_convert_kernel(const float* __restrict__ W, __bf16* __restrict__ Wt,
                                  int K, int N)
{
  int idx = blockIdx.x * 256 + threadIdx.x;
  if (idx >= K * N) return;
  int k = idx / N, n = idx % N;
  Wt[(size_t)n * K + k] = f2bf(W[idx]);
}

// Pack K as bf16 [b,h,s,d] and V transposed bf16 [b,h,d,s].
__global__ void qkv_pack_kernel(const float* __restrict__ kb, const float* __restrict__ vb,
                                __bf16* __restrict__ kbf, __bf16* __restrict__ vT)
{
  int idx = blockIdx.x * 256 + threadIdx.x;
  constexpr int total = Bc * NHc * Sc * DHc;
  if (idx >= total) return;
  int d = idx % DHc;
  int s = (idx / DHc) % Sc;
  int h = (idx / (DHc * Sc)) % NHc;
  int b = idx / (DHc * Sc * NHc);
  size_t src = ((size_t)(b * Sc + s)) * Hc + h * DHc + d;
  kbf[idx] = f2bf(kb[src]);
  vT[(((size_t)(b * NHc + h)) * DHc + d) * Sc + s] = f2bf(vb[src]);
}

// LayerNorm over H=768, fp32 out + optional bf16 copy.
__global__ __launch_bounds__(256) void ln_kernel(
    const float* __restrict__ x, float* __restrict__ y, __bf16* __restrict__ ybf,
    const float* __restrict__ g, const float* __restrict__ b)
{
  __shared__ float red[256];
  int t = blockIdx.x;
  int tid = threadIdx.x;
  const float* xr = x + (size_t)t * Hc;
  float e[3]; float sum = 0.f;
#pragma unroll
  for (int i = 0; i < 3; ++i) { e[i] = xr[tid + i * 256]; sum += e[i]; }
  red[tid] = sum; __syncthreads();
  for (int w = 128; w > 0; w >>= 1) { if (tid < w) red[tid] += red[tid + w]; __syncthreads(); }
  float mean = red[0] * (1.f / Hc);
  __syncthreads();
  float s2 = 0.f;
#pragma unroll
  for (int i = 0; i < 3; ++i) { float d = e[i] - mean; s2 += d * d; }
  red[tid] = s2; __syncthreads();
  for (int w = 128; w > 0; w >>= 1) { if (tid < w) red[tid] += red[tid + w]; __syncthreads(); }
  float rstd = rsqrtf(red[0] * (1.f / Hc) + 1e-12f);
#pragma unroll
  for (int i = 0; i < 3; ++i) {
    int j = tid + i * 256;
    float v = (e[i] - mean) * rstd * g[j] + b[j];
    y[(size_t)t * Hc + j] = v;
    if (ybf) ybf[(size_t)t * Hc + j] = f2bf(v);
  }
}

// Embeddings gather + LayerNorm + Bernoulli keep mask; fp32 + bf16 out.
__global__ __launch_bounds__(256) void embed_ln_kernel(
    const int* __restrict__ ids, const int* __restrict__ tok,
    const float* __restrict__ imp,
    const float* __restrict__ we, const float* __restrict__ pe,
    const float* __restrict__ te,
    const float* __restrict__ g, const float* __restrict__ bb,
    float* __restrict__ out, __bf16* __restrict__ outbf)
{
  __shared__ float red[256];
  int t = blockIdx.x;
  int s = t % Sc;
  int tid = threadIdx.x;
  int id = ids[t];
  int tt = tok[t];
  float e[3]; float sum = 0.f;
#pragma unroll
  for (int i = 0; i < 3; ++i) {
    int j = tid + i * 256;
    float v = we[(size_t)id * Hc + j] + pe[(size_t)s * Hc + j] + te[(size_t)tt * Hc + j];
    e[i] = v; sum += v;
  }
  red[tid] = sum; __syncthreads();
  for (int w = 128; w > 0; w >>= 1) { if (tid < w) red[tid] += red[tid + w]; __syncthreads(); }
  float mean = red[0] * (1.f / Hc);
  __syncthreads();
  float s2 = 0.f;
#pragma unroll
  for (int i = 0; i < 3; ++i) { float d = e[i] - mean; s2 += d * d; }
  red[tid] = s2; __syncthreads();
  for (int w = 128; w > 0; w >>= 1) { if (tid < w) red[tid] += red[tid + w]; __syncthreads(); }
  float rstd = rsqrtf(red[0] * (1.f / Hc) + 1e-12f);
  unsigned x = (unsigned)t * 2654435761u + 0x9E3779B9u;
  x ^= x >> 16; x *= 0x85EBCA6Bu; x ^= x >> 13; x *= 0xC2B2AE35u; x ^= x >> 16;
  float u = (float)x * 2.3283064e-10f;
  float keep = (u < imp[t]) ? 1.f : 0.f;
#pragma unroll
  for (int i = 0; i < 3; ++i) {
    int j = tid + i * 256;
    float v = ((e[i] - mean) * rstd * g[j] + bb[j]) * keep;
    out[(size_t)t * Hc + j]   = v;
    outbf[(size_t)t * Hc + j] = f2bf(v);
  }
}

__global__ void ext_mask_kernel(const float* __restrict__ mask, float* __restrict__ ext)
{
  int i = blockIdx.x * 256 + threadIdx.x;
  if (i < Bc * Sc) ext[i] = (1.f - mask[i]) * -10000.f;
}

// ---------------------------------------------------------------------------
extern "C" void kernel_launch(void* const* d_in, const int* in_sizes, int n_in,
                              void* d_out, int out_size, void* d_ws, size_t ws_size,
                              hipStream_t stream)
{
  const int*   input_ids = (const int*)  d_in[0];
  const float* attn_mask = (const float*)d_in[1];
  const int*   tok_type  = (const int*)  d_in[2];
  const float* imp       = (const float*)d_in[3];
  const float* word_emb  = (const float*)d_in[4];
  const float* pos_emb   = (const float*)d_in[5];
  const float* tok_emb   = (const float*)d_in[6];
  const float* emb_g     = (const float*)d_in[7];
  const float* emb_b     = (const float*)d_in[8];
  const float* Wq  = (const float*)d_in[9];
  const float* bq  = (const float*)d_in[10];
  const float* Wk  = (const float*)d_in[11];
  const float* bk  = (const float*)d_in[12];
  const float* Wv  = (const float*)d_in[13];
  const float* bv  = (const float*)d_in[14];
  const float* Wo  = (const float*)d_in[15];
  const float* bo  = (const float*)d_in[16];
  const float* g1  = (const float*)d_in[17];
  const float* b1  = (const float*)d_in[18];
  const float* Wi  = (const float*)d_in[19];
  const float* bi  = (const float*)d_in[20];
  const float* Wf  = (const float*)d_in[21];
  const float* bfv = (const float*)d_in[22];
  const float* g2  = (const float*)d_in[23];
  const float* b2  = (const float*)d_in[24];
  const float* Wp  = (const float*)d_in[25];
  const float* bp  = (const float*)d_in[26];

  // ---- workspace carve (~100 MB) ----
  char* ws = (char*)d_ws;
  auto carve = [&](size_t bytes) {
    char* p = ws; ws += (bytes + 255) & ~(size_t)255; return p;
  };
  float*  hbuf  = (float*) carve(sizeof(float) * Mtok * Hc);   // residual h
  float*  tmp   = (float*) carve(sizeof(float) * Mtok * Hc);   // pre-LN
  float*  aout  = (float*) carve(sizeof(float) * Mtok * Hc);   // attn_out (residual)
  float*  kbuf  = (float*) carve(sizeof(float) * Mtok * Hc);
  float*  vbuf  = (float*) carve(sizeof(float) * Mtok * Hc);
  float*  ctxb  = (float*) carve(sizeof(float) * Mtok * Hc);
  float*  extb  = (float*) carve(sizeof(float) * Bc * Sc);
  __bf16* wstg  = (__bf16*)carve(sizeof(__bf16) * Hc * FFc);                   // weight stage
  __bf16* hbf   = (__bf16*)carve(sizeof(__bf16) * Mtok * Hc);                  // bf16 h
  __bf16* abf   = (__bf16*)carve(sizeof(__bf16) * Mtok * Hc);                  // bf16 attn_out
  __bf16* ibf   = (__bf16*)carve(sizeof(__bf16) * Mtok * FFc);                 // bf16 inter
  __bf16* qbf   = (__bf16*)carve(sizeof(__bf16) * Mtok * Hc);                  // bf16 q
  __bf16* cbf   = (__bf16*)carve(sizeof(__bf16) * Mtok * Hc);                  // bf16 ctx
  __bf16* kbf   = (__bf16*)carve(sizeof(__bf16) * Bc * NHc * Sc * DHc);        // bf16 k
  __bf16* vT    = (__bf16*)carve(sizeof(__bf16) * Bc * NHc * Sc * DHc);        // bf16 v^T
  __bf16* pbf   = (__bf16*)carve(sizeof(__bf16) * (size_t)Bc * NHc * Sc * Sc); // bf16 probs

  float* out_seq    = (float*)d_out;                 // [B,S,H]
  float* out_pooled = out_seq + (size_t)Mtok * Hc;   // [B,H]
  float* out_attn   = out_pooled + (size_t)Bc * Hc;  // [L,B,NH,S,S]

  auto gemm_blocks = [](int M, int N) {
    return ((M + 127) / 128) * ((N + 127) / 128);
  };
  auto conv_blocks = [](int n) { return (n + 255) / 256; };

  embed_ln_kernel<<<Mtok, 256, 0, stream>>>(input_ids, tok_type, imp, word_emb,
                                            pos_emb, tok_emb, emb_g, emb_b, hbuf, hbf);
  ext_mask_kernel<<<conv_blocks(Bc * Sc), 256, 0, stream>>>(attn_mask, extb);

  for (int l = 0; l < Lc; ++l) {
    const float* Wql = Wq + (size_t)l * Hc * Hc;
    const float* Wkl = Wk + (size_t)l * Hc * Hc;
    const float* Wvl = Wv + (size_t)l * Hc * Hc;
    const float* Wol = Wo + (size_t)l * Hc * Hc;
    const float* Wil = Wi + (size_t)l * Hc * FFc;
    const float* Wfl = Wf + (size_t)l * FFc * Hc;
    float* attnL = out_attn + (size_t)l * Bc * NHc * Sc * Sc;

    // Q (bf16-only output), K, V projections
    wt_convert_kernel<<<conv_blocks(Hc * Hc), 256, 0, stream>>>(Wql, wstg, Hc, Hc);
    gemm_wmma_kernel<<<gemm_blocks(Mtok, Hc), 256, 0, stream>>>(
        hbf, Hc, wstg, Hc, nullptr, qbf, Hc, bq + l * Hc, nullptr, Mtok, Hc, Hc, 1.f, 0);
    wt_convert_kernel<<<conv_blocks(Hc * Hc), 256, 0, stream>>>(Wkl, wstg, Hc, Hc);
    gemm_wmma_kernel<<<gemm_blocks(Mtok, Hc), 256, 0, stream>>>(
        hbf, Hc, wstg, Hc, kbuf, nullptr, Hc, bk + l * Hc, nullptr, Mtok, Hc, Hc, 1.f, 0);
    wt_convert_kernel<<<conv_blocks(Hc * Hc), 256, 0, stream>>>(Wvl, wstg, Hc, Hc);
    gemm_wmma_kernel<<<gemm_blocks(Mtok, Hc), 256, 0, stream>>>(
        hbf, Hc, wstg, Hc, vbuf, nullptr, Hc, bv + l * Hc, nullptr, Mtok, Hc, Hc, 1.f, 0);

    // Attention
    qkv_pack_kernel<<<conv_blocks(Bc * NHc * Sc * DHc), 256, 0, stream>>>(kbuf, vbuf, kbf, vT);
    attn_scores_kernel<<<dim3(32, 1, Bc * NHc), 256, 0, stream>>>(qbf, kbf, extb, attnL);
    softmax_kernel<<<Bc * NHc * Sc, 256, 0, stream>>>(attnL, pbf);
    attn_ctx_kernel<<<dim3(4, 1, Bc * NHc), 256, 0, stream>>>(pbf, vT, ctxb, cbf);

    // Output projection + residual + LN1
    wt_convert_kernel<<<conv_blocks(Hc * Hc), 256, 0, stream>>>(Wol, wstg, Hc, Hc);
    gemm_wmma_kernel<<<gemm_blocks(Mtok, Hc), 256, 0, stream>>>(
        cbf, Hc, wstg, Hc, tmp, nullptr, Hc, bo + l * Hc, hbuf, Mtok, Hc, Hc, 1.f, 0);
    ln_kernel<<<Mtok, 256, 0, stream>>>(tmp, aout, abf, g1 + l * Hc, b1 + l * Hc);

    // FFN: GELU GEMM (bf16-only out), down GEMM + residual, LN2
    wt_convert_kernel<<<conv_blocks(Hc * FFc), 256, 0, stream>>>(Wil, wstg, Hc, FFc);
    gemm_wmma_kernel<<<gemm_blocks(Mtok, FFc), 256, 0, stream>>>(
        abf, Hc, wstg, Hc, nullptr, ibf, FFc, bi + l * FFc, nullptr, Mtok, FFc, Hc, 1.f, 1);
    wt_convert_kernel<<<conv_blocks(FFc * Hc), 256, 0, stream>>>(Wfl, wstg, FFc, Hc);
    gemm_wmma_kernel<<<gemm_blocks(Mtok, Hc), 256, 0, stream>>>(
        ibf, FFc, wstg, FFc, tmp, nullptr, Hc, bfv + l * Hc, aout, Mtok, Hc, FFc, 1.f, 0);
    float* lnout = (l == Lc - 1) ? out_seq : hbuf;
    ln_kernel<<<Mtok, 256, 0, stream>>>(tmp, lnout, hbf, g2 + l * Hc, b2 + l * Hc);
  }

  // Pooled output: tanh(seq_out[:,0] @ Wp + bp), M=4 (clamped loads, guarded stores).
  wt_convert_kernel<<<conv_blocks(Hc * Hc), 256, 0, stream>>>(Wp, wstg, Hc, Hc);
  gemm_wmma_kernel<<<gemm_blocks(Bc, Hc), 256, 0, stream>>>(
      hbf, Sc * Hc, wstg, Hc, out_pooled, nullptr, Hc, bp, nullptr, Bc, Hc, Hc, 1.f, 2);

  (void)in_sizes; (void)n_in; (void)out_size; (void)ws_size;
}
